// EndoWeightsExoGating_68384469287345
// MI455X (gfx1250) — matrix operand order repaired and checked by reading, not compiled, not run on hardware
//
#include <hip/hip_runtime.h>
#include <cmath>

// ---- problem constants ----
#define B_   2
#define TQ_  256
#define D_   256
#define N_   128
#define TK_  128
#define H_   8
#define HD_  32
#define SK_  (N_*TK_)   // 16384

typedef __attribute__((ext_vector_type(16))) __bf16       v16bf;
typedef __attribute__((ext_vector_type(8)))  float        v8f;
typedef __attribute__((ext_vector_type(4)))  unsigned int v4u;

// float -> bf16 (round-to-nearest-even) without relying on __bf16 arithmetic
__device__ __forceinline__ __bf16 f2bf(float f) {
  union { float f; unsigned u; } x; x.f = f;
  unsigned r = (x.u + 0x7FFFu + ((x.u >> 16) & 1u)) >> 16;
  union { unsigned short s; __bf16 b; } y; y.s = (unsigned short)r;
  return y.b;
}

// A fragment (16x32 bf16): lane m=lane&15, half=lane>>4; dwords {0..3}+4h and {8..11}+4h of row m
__device__ __forceinline__ v16bf load_frag_a(const __bf16* base, int stride) {
  const int lane = threadIdx.x & 31;
  const int m = lane & 15, hf = lane >> 4;
  const v4u* p = (const v4u*)(base + (size_t)m * stride);
  union { v16bf v; v4u q[2]; } r;
  r.q[0] = p[hf];
  r.q[1] = p[2 + hf];
  return r.v;
}

// B fragment (32x16 bf16) sourced from "rows indexed by N, K contiguous" memory:
// lane n=lane&15, half=lane>>4; dwords {0..3}+4h and {8..11}+4h of row n (same K order as A)
__device__ __forceinline__ v16bf load_frag_b(const __bf16* base, int stride) {
  const int lane = threadIdx.x & 31;
  const int nn = lane & 15, hf = lane >> 4;
  const v4u* p = (const v4u*)(base + (size_t)nn * stride);
  union { v16bf v; v4u q[2]; } r;
  r.q[0] = p[hf];
  r.q[1] = p[2 + hf];
  return r.v;
}

__device__ __forceinline__ v8f wmma_bf16(v16bf a, v16bf b, v8f c) {
  return __builtin_amdgcn_wmma_f32_16x16x32_bf16(false, a, false, b, (short)0, c, false, false);
}

__device__ __forceinline__ float wsum32(float v) {
#pragma unroll
  for (int m = 1; m < 32; m <<= 1) v += __shfl_xor(v, m, 32);
  return v;
}

// ---------------- prep kernels ----------------
__global__ void convert_bf16_kernel(const float* __restrict__ src, __bf16* __restrict__ dst, int n) {
  for (int i = blockIdx.x * blockDim.x + threadIdx.x; i < n; i += gridDim.x * blockDim.x)
    dst[i] = f2bf(src[i]);
}

// dst[c*rows + r] = src[r*cols + c]
__global__ void transpose_bf16_kernel(const float* __restrict__ src, __bf16* __restrict__ dst,
                                      int rows, int cols) {
  int n = rows * cols;
  for (int i = blockIdx.x * blockDim.x + threadIdx.x; i < n; i += gridDim.x * blockDim.x) {
    int r = i / cols, c = i - r * cols;
    dst[(size_t)c * rows + r] = f2bf(src[i]);
  }
}

// exo[b][n][s][d] -> exo_t[b*n][d][s] (bf16)
__global__ void exo_transpose_kernel(const float* __restrict__ exo, __bf16* __restrict__ dst) {
  const int n = B_ * N_ * TK_ * D_;
  for (int i = blockIdx.x * blockDim.x + threadIdx.x; i < n; i += gridDim.x * blockDim.x) {
    int bn = i / (TK_ * D_);
    int rem = i - bn * (TK_ * D_);
    int s = rem >> 8;           // / D_
    int d = rem & (D_ - 1);
    dst[((size_t)bn * D_ + d) * TK_ + s] = f2bf(exo[i]);
  }
}

// fused[b][0][t][d] = endo[b][t][d]
__global__ void copy_endo_kernel(const float* __restrict__ endo, float* __restrict__ fused) {
  const int n = B_ * TQ_ * D_;
  for (int i = blockIdx.x * blockDim.x + threadIdx.x; i < n; i += gridDim.x * blockDim.x) {
    int b = i / (TQ_ * D_);
    int rem = i - b * (TQ_ * D_);
    fused[(size_t)b * 129 * TQ_ * D_ + rem] = endo[i];
  }
}

// ---------------- projection GEMM: out[rows x D] = A[rows x D] * W^T + bias ----------------
// W stored row-major [dout][din] (so B-fragment rows are W rows). out stored bf16.
__global__ void proj_gemm_kernel(const __bf16* __restrict__ A, const __bf16* __restrict__ W,
                                 const float* __restrict__ bias, __bf16* __restrict__ out, int rows) {
  const int w = threadIdx.x >> 5, lane = threadIdx.x & 31;
  const int nn = lane & 15, hf = lane >> 4;
  const int tile = blockIdx.x * (blockDim.x >> 5) + w;
  const int ntiles = (rows >> 4) * (D_ >> 4);
  if (tile >= ntiles) return;
  const int mi = tile >> 4;     // D_/16 == 16 col tiles
  const int ni = tile & 15;
  const __bf16* abase = A + (size_t)mi * 16 * D_;
  const __bf16* bbase = W + (size_t)ni * 16 * D_;
  v8f c = {};
#pragma unroll
  for (int kk = 0; kk < 8; ++kk) {
    v16bf a = load_frag_a(abase + kk * 32, D_);
    v16bf b = load_frag_b(bbase + kk * 32, D_);
    c = wmma_bf16(a, b, c);
  }
  const int ncol = ni * 16 + nn;
  const float bv = bias[ncol];
#pragma unroll
  for (int r = 0; r < 8; ++r)
    out[((size_t)(mi * 16 + r + 8 * hf)) * D_ + ncol] = f2bf(c[r] + bv);
}

// ---------------- attention block-sums -> beta/gamma ----------------
// grid = B_*(TQ_/16) blocks of 256 threads (8 waves = 8 heads).
// Two streaming passes over SK with WMMA scores; never materializes attn weights.
__global__ void attn_beta_kernel(const __bf16* __restrict__ q_bf, const __bf16* __restrict__ k_bf,
                                 float* __restrict__ beta_out, float* __restrict__ gamma_out,
                                 __bf16* __restrict__ beta_bf) {
  __shared__ float svals[16][N_];    // 8KB: s[t_local][nblock], head-averaged
  for (int i = threadIdx.x; i < 16 * N_; i += blockDim.x) (&svals[0][0])[i] = 0.f;
  __syncthreads();

  const int b  = blockIdx.x >> 4;
  const int qt = blockIdx.x & 15;
  const int h  = threadIdx.x >> 5;
  const int lane = threadIdx.x & 31;
  const int nn = lane & 15, hf = lane >> 4;
  const float scale = 0.17677669529663687f;   // 1/sqrt(32)

  const __bf16* qbase = q_bf + ((size_t)b * TQ_ + qt * 16) * D_ + h * HD_;
  const v16bf afrag = load_frag_a(qbase, D_);
  const __bf16* kcol = k_bf + (size_t)b * SK_ * D_ + h * HD_;

  float rm[8], zz[8];
#pragma unroll
  for (int r = 0; r < 8; ++r) { rm[r] = -1e30f; zz[r] = 0.f; }

  // pass 1: online max + Z per lane (over this lane's score columns)
  for (int st = 0; st < SK_ / 16; ++st) {
    __builtin_prefetch(kcol + (size_t)(st + 8) * 16 * D_, 0, 1);
    v16bf bfrag = load_frag_b(kcol + (size_t)st * 16 * D_, D_);
    v8f c = {};
    c = wmma_bf16(afrag, bfrag, c);
#pragma unroll
    for (int r = 0; r < 8; ++r) {
      float s = c[r];
      float nm = fmaxf(rm[r], s);
      zz[r] = zz[r] * __expf((rm[r] - nm) * scale) + __expf((s - nm) * scale);
      rm[r] = nm;
    }
  }
  // combine (max, Z) across the 16 lanes sharing a half (masks < 16 keep lanes in-group)
#pragma unroll
  for (int r = 0; r < 8; ++r) {
#pragma unroll
    for (int mask = 1; mask < 16; mask <<= 1) {
      float om = __shfl_xor(rm[r], mask, 32);
      float oz = __shfl_xor(zz[r], mask, 32);
      float nm = fmaxf(rm[r], om);
      zz[r] = zz[r] * __expf((rm[r] - nm) * scale) + oz * __expf((om - nm) * scale);
      rm[r] = nm;
    }
  }
  float invz[8];
#pragma unroll
  for (int r = 0; r < 8; ++r) invz[r] = 1.f / ((float)H_ * fmaxf(zz[r], 1e-30f));

  // pass 2: per-(row, 128-key-block) exp sums, averaged over heads via shared atomics
  float acc[8];
#pragma unroll
  for (int r = 0; r < 8; ++r) acc[r] = 0.f;
  for (int st = 0; st < SK_ / 16; ++st) {
    __builtin_prefetch(kcol + (size_t)(st + 8) * 16 * D_, 0, 1);
    v16bf bfrag = load_frag_b(kcol + (size_t)st * 16 * D_, D_);
    v8f c = {};
    c = wmma_bf16(afrag, bfrag, c);
#pragma unroll
    for (int r = 0; r < 8; ++r) acc[r] += __expf((c[r] - rm[r]) * scale);
    if ((st & 7) == 7) {                 // one TK_=128 block finished
      const int nb = st >> 3;
#pragma unroll
      for (int r = 0; r < 8; ++r) {
        float bs = acc[r];
#pragma unroll
        for (int mask = 1; mask < 16; mask <<= 1) bs += __shfl_xor(bs, mask, 32);
        if (nn == 0) atomicAdd(&svals[r + 8 * hf][nb], bs * invz[r]);
        acc[r] = 0.f;
      }
    }
  }
  __syncthreads();

  // row-normalize (safe) and emit beta, gamma (f32) + beta_bf (bf16 GEMM operand)
#pragma unroll
  for (int j = 0; j < 2; ++j) {
    const int m = h * 2 + j;
    float p = 0.f;
    for (int i = lane; i < N_; i += 32) p += svals[m][i];
    p = wsum32(p);
    const float denom = fmaxf(p, 1e-9f);
    const int t = qt * 16 + m;
    const size_t roff = ((size_t)b * TQ_ + t) * N_;
    for (int i = lane; i < N_; i += 32) {
      float v = svals[m][i] / denom;
      if (!(v == v) || fabsf(v) > 1e30f) v = 0.f;
      beta_out[roff + i]  = v;
      gamma_out[roff + i] = v;
      beta_bf[roff + i]   = f2bf(v);
    }
  }
}

// ---------------- exo mixing + gate + LN1 ----------------
// grid = B_*N_*4 blocks of 128 threads (4 waves); each block: one (b,n), 64 t-rows.
__global__ void mix_ln1_kernel(const __bf16* __restrict__ beta_bf, const __bf16* __restrict__ exo_t,
                               const float* __restrict__ gamma, const float* __restrict__ alpha_p,
                               const float* __restrict__ ln1_g, const float* __restrict__ ln1_b,
                               float* __restrict__ x_f32, __bf16* __restrict__ x_bf) {
  __shared__ float xm[64][D_];   // 64KB
  const int bid = blockIdx.x;
  const int b = bid >> 9;                 // / (N_*4)
  const int rem = bid & 511;
  const int n = rem >> 2;
  const int tq = rem & 3;
  const int w = threadIdx.x >> 5, lane = threadIdx.x & 31;
  const int nn = lane & 15, hf = lane >> 4;
  const float alpha = *alpha_p;
  const int t0 = tq * 64 + w * 16;

  const __bf16* abase = beta_bf + ((size_t)b * TQ_ + t0) * N_;
  v16bf a0 = load_frag_a(abase +  0, N_);
  v16bf a1 = load_frag_a(abase + 32, N_);
  v16bf a2 = load_frag_a(abase + 64, N_);
  v16bf a3 = load_frag_a(abase + 96, N_);

  float gf[8];
#pragma unroll
  for (int r = 0; r < 8; ++r) {
    const int t = t0 + r + 8 * hf;
    const float g = gamma[((size_t)b * TQ_ + t) * N_ + n];
    gf[r] = 1.f + alpha * (g - 1.f);      // exo_upd = exo_mixed*(1+alpha*(g-1))
  }

  const __bf16* bt = exo_t + (size_t)(b * N_ + n) * D_ * TK_;
  for (int dt = 0; dt < 16; ++dt) {
    const __bf16* bb = bt + (size_t)dt * 16 * TK_;
    v8f c = {};
    c = wmma_bf16(a0, load_frag_b(bb +  0, TK_), c);
    c = wmma_bf16(a1, load_frag_b(bb + 32, TK_), c);
    c = wmma_bf16(a2, load_frag_b(bb + 64, TK_), c);
    c = wmma_bf16(a3, load_frag_b(bb + 96, TK_), c);
#pragma unroll
    for (int r = 0; r < 8; ++r)
      xm[w * 16 + r + 8 * hf][dt * 16 + nn] = c[r] * gf[r];
  }
  __syncthreads();

  // LayerNorm over D per row; each wave handles its own 16 rows
  for (int j = 0; j < 16; ++j) {
    const int lr = w * 16 + j;
    const int t = tq * 64 + lr;
    float v[8], s = 0.f;
#pragma unroll
    for (int i = 0; i < 8; ++i) { v[i] = xm[lr][lane + 32 * i]; s += v[i]; }
    s = wsum32(s);
    const float mu = s * (1.f / D_);
    float var = 0.f;
#pragma unroll
    for (int i = 0; i < 8; ++i) { float d = v[i] - mu; var += d * d; }
    var = wsum32(var) * (1.f / D_);
    const float inv = rsqrtf(var + 1e-5f);
    const size_t roff = ((size_t)(b * N_ + n) * TQ_ + t) * D_;
#pragma unroll
    for (int i = 0; i < 8; ++i) {
      const int d = lane + 32 * i;
      const float y = (v[i] - mu) * inv * ln1_g[d] + ln1_b[d];
      x_f32[roff + d] = y;
      x_bf[roff + d]  = f2bf(y);
    }
  }
}

// ---------------- fused FFN + residual + LN2 -> fused[:,1:] ----------------
// grid = 4096 blocks of 256 threads (8 waves); each block: 16 rows through LDS.
__global__ void ffn_kernel(const __bf16* __restrict__ x_bf, const float* __restrict__ x_f32,
                           const __bf16* __restrict__ w1t, const float* __restrict__ b1,
                           const __bf16* __restrict__ w2t, const float* __restrict__ b2,
                           const float* __restrict__ ln2_g, const float* __restrict__ ln2_b,
                           float* __restrict__ fused) {
  __shared__ __bf16 xs[16][D_];        // 8KB
  __shared__ __bf16 hs[16][4 * D_];    // 32KB
  __shared__ float  ys[16][D_];        // 16KB
  const int r0 = blockIdx.x * 16;
  const int w = threadIdx.x >> 5, lane = threadIdx.x & 31;
  const int nn = lane & 15, hf = lane >> 4;

  { // stage x rows into LDS with CDNA5 async copies: memory -> LDS directly,
    // no VGPR round-trip, tracked by ASYNCcnt (8KB = 512 x b128 transfers).
    unsigned ldsbase = (unsigned)(size_t)(void*)&xs[0][0];
    const char* gbase = (const char*)(x_bf + (size_t)r0 * D_);
#pragma unroll
    for (int it = 0; it < 2; ++it) {
      const int idx = threadIdx.x + it * 256;
      unsigned lo = ldsbase + idx * 16;
      const char* ga = gbase + (size_t)idx * 16;
      asm volatile("global_load_async_to_lds_b128 %0, %1, off"
                   :: "v"(lo), "v"(ga) : "memory");
    }
    asm volatile("s_wait_asynccnt 0x0" ::: "memory");
  }
  __syncthreads();

  // GEMM1 (16x256 @ 256x1024) + bias + exact GeLU -> hs
  v16bf afr[8];
#pragma unroll
  for (int kk = 0; kk < 8; ++kk) afr[kk] = load_frag_a(&xs[0][kk * 32], D_);
  for (int tn = 0; tn < 8; ++tn) {
    const int ni = w * 8 + tn;
    v8f c = {};
#pragma unroll
    for (int kk = 0; kk < 8; ++kk)
      c = wmma_bf16(afr[kk], load_frag_b(w1t + (size_t)(ni * 16) * D_ + kk * 32, D_), c);
    const int ncol = ni * 16 + nn;
    const float bias = b1[ncol];
#pragma unroll
    for (int r = 0; r < 8; ++r) {
      float v = c[r] + bias;
      v = 0.5f * v * (1.f + erff(v * 0.70710678118654752f));
      hs[r + 8 * hf][ncol] = f2bf(v);
    }
  }
  __syncthreads();

  // GEMM2 (16x1024 @ 1024x256) + bias + residual -> ys
  for (int tn = 0; tn < 2; ++tn) {
    const int ni = w * 2 + tn;
    v8f c = {};
    for (int kk = 0; kk < 32; ++kk) {
      v16bf a = load_frag_a(&hs[0][kk * 32], 4 * D_);
      v16bf b = load_frag_b(w2t + (size_t)(ni * 16) * (4 * D_) + kk * 32, 4 * D_);
      c = wmma_bf16(a, b, c);
    }
    const int ncol = ni * 16 + nn;
    const float bias = b2[ncol];
#pragma unroll
    for (int r = 0; r < 8; ++r) {
      const int row = r + 8 * hf;
      ys[row][ncol] = c[r] + bias + x_f32[(size_t)(r0 + row) * D_ + ncol];
    }
  }
  __syncthreads();

  // LN2 + scatter into fused[b][1+n][t][:]
#pragma unroll
  for (int j = 0; j < 2; ++j) {
    const int lr = w * 2 + j;
    const int gr = r0 + lr;
    const int b = gr / (N_ * TQ_);
    const int rem = gr - b * (N_ * TQ_);
    const int n = rem / TQ_;
    const int t = rem - n * TQ_;
    float v[8], s = 0.f;
#pragma unroll
    for (int i = 0; i < 8; ++i) { v[i] = ys[lr][lane + 32 * i]; s += v[i]; }
    s = wsum32(s);
    const float mu = s * (1.f / D_);
    float var = 0.f;
#pragma unroll
    for (int i = 0; i < 8; ++i) { float d = v[i] - mu; var += d * d; }
    var = wsum32(var) * (1.f / D_);
    const float inv = rsqrtf(var + 1e-5f);
    const size_t off = (((size_t)b * 129 + 1 + n) * TQ_ + t) * (size_t)D_;
#pragma unroll
    for (int i = 0; i < 8; ++i) {
      const int d = lane + 32 * i;
      fused[off + d] = (v[i] - mu) * inv * ln2_g[d] + ln2_b[d];
    }
  }
}

// ---------------- launch ----------------
extern "C" void kernel_launch(void* const* d_in, const int* in_sizes, int n_in,
                              void* d_out, int out_size, void* d_ws, size_t ws_size,
                              hipStream_t stream) {
  const float* endo      = (const float*)d_in[0];
  const float* exo       = (const float*)d_in[1];
  const float* in_proj_w = (const float*)d_in[2];
  const float* in_proj_b = (const float*)d_in[3];
  const float* alpha     = (const float*)d_in[4];
  const float* ln1_g     = (const float*)d_in[5];
  const float* ln1_b     = (const float*)d_in[6];
  const float* w1        = (const float*)d_in[7];
  const float* b1        = (const float*)d_in[8];
  const float* w2        = (const float*)d_in[9];
  const float* b2        = (const float*)d_in[10];
  const float* ln2_g     = (const float*)d_in[11];
  const float* ln2_b     = (const float*)d_in[12];

  float* fused     = (float*)d_out;
  float* beta_out  = fused + (size_t)B_ * 129 * TQ_ * D_;
  float* gamma_out = beta_out + (size_t)B_ * TQ_ * N_;

  char* ws = (char*)d_ws;
  __bf16* wqk_bf  = (__bf16*)(ws + 0);          // [512][256]  (Wq rows then Wk rows)
  __bf16* w1t_bf  = (__bf16*)(ws + 262144);     // [1024][256] = w1^T
  __bf16* w2t_bf  = (__bf16*)(ws + 786432);     // [256][1024] = w2^T
  __bf16* endo_bf = (__bf16*)(ws + 1310720);
  __bf16* q_bf    = (__bf16*)(ws + 1572864);    // [B][TQ][D]
  __bf16* beta_bf = (__bf16*)(ws + 1835008);    // [B][TQ][N]
  __bf16* k_bf    = (__bf16*)(ws + 2097152);    // [B][SK][D]
  __bf16* exo_bf  = (__bf16*)(ws + 18874368);   // [B][N][TK][D]
  __bf16* exo_t   = (__bf16*)(ws + 52428800);   // [B*N][D][TK]
  float*  x_f32   = (float*) (ws + 85983232);   // [B][N][TQ][D]
  __bf16* x_bf    = (__bf16*)(ws + 153092096);  // [B][N][TQ][D]

  convert_bf16_kernel<<<512, 256, 0, stream>>>(in_proj_w, wqk_bf, 2 * D_ * D_);
  convert_bf16_kernel<<<512, 256, 0, stream>>>(endo, endo_bf, B_ * TQ_ * D_);
  convert_bf16_kernel<<<8192, 256, 0, stream>>>(exo, exo_bf, B_ * N_ * TK_ * D_);
  transpose_bf16_kernel<<<1024, 256, 0, stream>>>(w1, w1t_bf, D_, 4 * D_);
  transpose_bf16_kernel<<<1024, 256, 0, stream>>>(w2, w2t_bf, 4 * D_, D_);
  exo_transpose_kernel<<<8192, 256, 0, stream>>>(exo, exo_t);
  copy_endo_kernel<<<512, 256, 0, stream>>>(endo, fused);

  proj_gemm_kernel<<<64, 256, 0, stream>>>(endo_bf, wqk_bf, in_proj_b, q_bf, B_ * TQ_);
  proj_gemm_kernel<<<4096, 256, 0, stream>>>(exo_bf, wqk_bf + D_ * D_, in_proj_b + D_, k_bf, B_ * SK_);

  attn_beta_kernel<<<B_ * (TQ_ / 16), 256, 0, stream>>>(q_bf, k_bf, beta_out, gamma_out, beta_bf);

  mix_ln1_kernel<<<B_ * N_ * 4, 128, 0, stream>>>(beta_bf, exo_t, gamma_out, alpha,
                                                  ln1_g, ln1_b, x_f32, x_bf);

  ffn_kernel<<<(B_ * N_ * TQ_) / 16, 256, 0, stream>>>(x_bf, x_f32, w1t_bf, b1, w2t_bf, b2,
                                                       ln2_g, ln2_b, fused);
}
